// MixtureOfExperts_63531156242852
// MI455X (gfx1250) — matrix operand order repaired and checked by reading, not compiled, not run on hardware
//
#include <hip/hip_runtime.h>

#define T_TOK 4096   // B*S tokens
#define HD    1024   // hidden
#define ID    4096   // intermediate
#define NE    8      // experts
#define MBLK  32     // tokens per workgroup
#define IC    256    // intermediate chunk
#define XS    (HD + 8)   // padded LDS row stride (bf16 elems) for X
#define AS    (IC + 8)   // padded LDS row stride for act chunk

typedef __attribute__((ext_vector_type(16))) __bf16 v16bf;
typedef __attribute__((ext_vector_type(8)))  __bf16 v8bf;
typedef __attribute__((ext_vector_type(8)))  float  v8f;

// ---------- bf16 helpers (RNE, bit-exact) --------------------------------------
static __device__ __forceinline__ __bf16 f2bf(float f) {
    unsigned u = __builtin_bit_cast(unsigned, f);
    u += 0x7fffu + ((u >> 16) & 1u);
    unsigned short s = (unsigned short)(u >> 16);
    return __builtin_bit_cast(__bf16, s);
}
static __device__ __forceinline__ float bf2f(__bf16 b) {
    unsigned short s = __builtin_bit_cast(unsigned short, b);
    unsigned u = ((unsigned)s) << 16;
    return __builtin_bit_cast(float, u);
}

// ---------- WMMA helpers -------------------------------------------------------
// D = C + Ahi*Bhi + Alo*Bhi + Ahi*Blo   (bf16x3, f32 accumulate)
static __device__ __forceinline__ v8f wmma3(v16bf ah, v16bf al, v16bf bh, v16bf bl, v8f c) {
    c = __builtin_amdgcn_wmma_f32_16x16x32_bf16(false, ah, false, bh, (short)0, c, false, false);
    c = __builtin_amdgcn_wmma_f32_16x16x32_bf16(false, al, false, bh, (short)0, c, false, false);
    c = __builtin_amdgcn_wmma_f32_16x16x32_bf16(false, ah, false, bl, (short)0, c, false, false);
    return c;
}

// A operand (16x32 bf16) from LDS, row-major padded rows.
static __device__ __forceinline__ v16bf load_a(const __bf16* s, int stride, int m0, int k0, int lane) {
    const int half = lane >> 4;
    const int r    = m0 + (lane & 15);
    const __bf16* p = s + r * stride + k0 + half * 16;
    v8bf x0 = *(const v8bf*)(p);
    v8bf x1 = *(const v8bf*)(p + 8);
    return __builtin_shufflevector(x0, x1, 0,1,2,3,4,5,6,7,8,9,10,11,12,13,14,15);
}

// B operand from pre-swizzled weights: tile = 32 lanes x 16 bf16 contiguous.
// Two coalesced 16B loads per lane, no conversion, no layout shuffle.
static __device__ __forceinline__ v16bf load_swz(const __bf16* __restrict__ arr,
                                                 size_t tile, int lane) {
    const __bf16* p = arr + (tile * 32 + (size_t)lane) * 16;
    v8bf x0 = *(const v8bf*)(p);
    v8bf x1 = *(const v8bf*)(p + 8);
    return __builtin_shufflevector(x0, x1, 0,1,2,3,4,5,6,7,8,9,10,11,12,13,14,15);
}

// ---------- kernel 0: weight convert + swizzle pre-pass ------------------------
// src: [Ktot][Ntot] f32 (expert dim folded into Ktot; 32 | Ktot per expert).
// For tile (kt,nt): lane L holds W[kt*32 + (L>>4)*16 + i][nt*16 + (L&15)], i=0..15,
// stored contiguously -> contiguous k order, same as load_a's A operand.
__global__ void moe_swz_kernel(const float* __restrict__ src,
                               __bf16* __restrict__ hi, __bf16* __restrict__ lo,
                               int Ntot, int ntiles_n, int total_tiles) {
    const int tile = blockIdx.x * 8 + (threadIdx.x >> 5);
    if (tile >= total_tiles) return;
    const int lane = threadIdx.x & 31;
    const int kt = tile / ntiles_n;
    const int nt = tile % ntiles_n;
    const int n  = nt * 16 + (lane & 15);
    const int kb = kt * 32 + (lane >> 4) * 16;
    const float* p = src + (size_t)kb * Ntot + n;
    v8bf h0, h1, l0, l1;
#pragma unroll
    for (int i = 0; i < 8; ++i) {
        float f = p[(size_t)i * Ntot];
        __bf16 h = f2bf(f);
        h0[i] = h; l0[i] = f2bf(f - bf2f(h));
    }
#pragma unroll
    for (int i = 0; i < 8; ++i) {
        float f = p[(size_t)(i + 8) * Ntot];
        __bf16 h = f2bf(f);
        h1[i] = h; l1[i] = f2bf(f - bf2f(h));
    }
    __bf16* ph = hi + ((size_t)tile * 32 + lane) * 16;
    __bf16* pl = lo + ((size_t)tile * 32 + lane) * 16;
    *(v8bf*)(ph)     = h0;
    *(v8bf*)(ph + 8) = h1;
    *(v8bf*)(pl)     = l0;
    *(v8bf*)(pl + 8) = l1;
}

// ---------- kernel 1: zero the 'out' region -----------------------------------
__global__ void moe_zero_kernel(float4* __restrict__ p, int n4) {
    int i = blockIdx.x * blockDim.x + threadIdx.x;
    if (i < n4) p[i] = make_float4(0.f, 0.f, 0.f, 0.f);
}

// ---------- kernel 2: router (one wave per token) ------------------------------
__global__ void moe_router_kernel(const float* __restrict__ hs,
                                  const float* __restrict__ rw,
                                  float* __restrict__ logits_out,
                                  float* __restrict__ sel_out,
                                  float* __restrict__ tok_wt,
                                  int*   __restrict__ tok_sel) {
    const int wv   = threadIdx.x >> 5;
    const int lane = threadIdx.x & 31;
    const int t    = blockIdx.x * 8 + wv;
    if (t >= T_TOK) return;

    float acc[NE];
#pragma unroll
    for (int e = 0; e < NE; ++e) acc[e] = 0.f;

    const float* xr = hs + (size_t)t * HD;
    for (int h = lane; h < HD; h += 32) {
        float x = xr[h];
        const float* r = rw + (size_t)h * NE;
#pragma unroll
        for (int e = 0; e < NE; ++e) acc[e] += x * r[e];
    }
#pragma unroll
    for (int off = 16; off > 0; off >>= 1) {
#pragma unroll
        for (int e = 0; e < NE; ++e) acc[e] += __shfl_xor(acc[e], off, 32);
    }

    if (lane == 0) {
        float mx = acc[0];
#pragma unroll
        for (int e = 1; e < NE; ++e) mx = acc[e] > mx ? acc[e] : mx;
        float p[NE];
#pragma unroll
        for (int e = 0; e < NE; ++e) p[e] = __expf(acc[e] - mx);
        int e0 = 0;
#pragma unroll
        for (int e = 1; e < NE; ++e) if (p[e] > p[e0]) e0 = e;
        int e1 = (e0 == 0) ? 1 : 0;
#pragma unroll
        for (int e = 0; e < NE; ++e) if (e != e0 && p[e] > p[e1]) e1 = e;
        float s = p[e0] + p[e1];
        float w0 = p[e0] / s, w1 = p[e1] / s;
#pragma unroll
        for (int e = 0; e < NE; ++e) logits_out[(size_t)t * NE + e] = acc[e];
        sel_out[(size_t)t * 2 + 0] = (float)e0;
        sel_out[(size_t)t * 2 + 1] = (float)e1;
        tok_sel[t * 2 + 0] = e0;
        tok_sel[t * 2 + 1] = e1;
        tok_wt[t * 2 + 0] = w0;
        tok_wt[t * 2 + 1] = w1;
    }
}

// ---------- kernel 3: deterministic per-expert compaction ----------------------
__global__ void moe_compact_kernel(const int* __restrict__ tok_sel,
                                   const float* __restrict__ tok_wt,
                                   int* __restrict__ etok, float* __restrict__ ewt,
                                   int* __restrict__ ecnt) {
    int e = threadIdx.x;
    if (e >= NE) return;
    int c = 0;
    for (int t = 0; t < T_TOK; ++t) {
#pragma unroll
        for (int j = 0; j < 2; ++j) {
            if (tok_sel[t * 2 + j] == e) {
                etok[e * T_TOK + c] = t;
                ewt[e * T_TOK + c]  = tok_wt[t * 2 + j];
                ++c;
            }
        }
    }
    ecnt[e] = c;
}

// ---------- kernel 4: fused SwiGLU expert FFN (WMMA, bf16x3) -------------------
__global__ __launch_bounds__(256, 1)
void moe_ffn_kernel(const float* __restrict__ hs,
                    const __bf16* __restrict__ wg_h, const __bf16* __restrict__ wg_l,
                    const __bf16* __restrict__ wu_h, const __bf16* __restrict__ wu_l,
                    const __bf16* __restrict__ wd_h, const __bf16* __restrict__ wd_l,
                    const int*   __restrict__ ecnt,
                    const int*   __restrict__ etok,
                    const float* __restrict__ ewt,
                    float* __restrict__ out) {
    __shared__ __align__(16) __bf16 Xh[MBLK * XS];
    __shared__ __align__(16) __bf16 Xl[MBLK * XS];
    __shared__ __align__(16) __bf16 Ach[MBLK * AS];
    __shared__ __align__(16) __bf16 Acl[MBLK * AS];
    __shared__ int   s_tok[MBLK];
    __shared__ float s_wgt[MBLK];

    const int e   = blockIdx.x / (T_TOK / MBLK);
    const int mb  = blockIdx.x % (T_TOK / MBLK);
    const int cnt = ecnt[e];
    const int m0g = mb * MBLK;
    if (m0g >= cnt) return;            // uniform: EXEC stays all-ones for WMMA

    const int tid  = threadIdx.x;
    const int wv   = tid >> 5;         // wave 0..7
    const int lane = tid & 31;
    const int half = lane >> 4;
    const int ln   = lane & 15;

    if (tid < MBLK) {
        int idx = m0g + tid;
        int cl  = idx < cnt ? idx : cnt - 1;
        s_tok[tid] = etok[e * T_TOK + cl];
        s_wgt[tid] = idx < cnt ? ewt[e * T_TOK + cl] : 0.0f;
    }
    __syncthreads();

    // stage X block [32 x 1024] as bf16 hi/lo (coalesced float4 gather)
    for (int it = tid; it < MBLK * (HD / 4); it += 256) {
        int row = it >> 8;             // HD/4 == 256
        int c4  = it & 255;
        const float4 v = *(const float4*)(hs + (size_t)s_tok[row] * HD + c4 * 4);
        float f[4] = {v.x, v.y, v.z, v.w};
#pragma unroll
        for (int j = 0; j < 4; ++j) {
            __bf16 h = f2bf(f[j]);
            Xh[row * XS + c4 * 4 + j] = h;
            Xl[row * XS + c4 * 4 + j] = f2bf(f[j] - bf2f(h));
        }
    }
    __syncthreads();

    v8f Y[2][8];
#pragma unroll
    for (int mt = 0; mt < 2; ++mt)
#pragma unroll
        for (int nt = 0; nt < 8; ++nt)
#pragma unroll
            for (int r = 0; r < 8; ++r) Y[mt][nt][r] = 0.f;

#pragma unroll 1
    for (int ci = 0; ci < ID / IC; ++ci) {
        // ---- phase A: G = X*Wg, U = X*Wu over this wave's 32 chunk columns ----
        v8f Cg[2][2], Cu[2][2];
#pragma unroll
        for (int mt = 0; mt < 2; ++mt)
#pragma unroll
            for (int nt = 0; nt < 2; ++nt)
#pragma unroll
                for (int r = 0; r < 8; ++r) { Cg[mt][nt][r] = 0.f; Cu[mt][nt][r] = 0.f; }

#pragma unroll 1
        for (int kt = 0; kt < HD / 32; ++kt) {
            const int k0 = kt * 32;
            v16bf a_h[2], a_l[2];
#pragma unroll
            for (int mt = 0; mt < 2; ++mt) {
                a_h[mt] = load_a(Xh, XS, mt * 16, k0, lane);
                a_l[mt] = load_a(Xl, XS, mt * 16, k0, lane);
            }
            // tile row index inside the [E*HD][ID] swizzled matrix
            const size_t ktg = (size_t)(e * (HD / 32) + kt);
#pragma unroll
            for (int nt = 0; nt < 2; ++nt) {
                const size_t tile = ktg * (ID / 16) + (ci * (IC / 16) + wv * 2 + nt);
                v16bf bh = load_swz(wg_h, tile, lane);
                v16bf bl = load_swz(wg_l, tile, lane);
                Cg[0][nt] = wmma3(a_h[0], a_l[0], bh, bl, Cg[0][nt]);
                Cg[1][nt] = wmma3(a_h[1], a_l[1], bh, bl, Cg[1][nt]);
                bh = load_swz(wu_h, tile, lane);
                bl = load_swz(wu_l, tile, lane);
                Cu[0][nt] = wmma3(a_h[0], a_l[0], bh, bl, Cu[0][nt]);
                Cu[1][nt] = wmma3(a_h[1], a_l[1], bh, bl, Cu[1][nt]);
            }
        }

        __syncthreads();   // previous chunk's phase-B readers are done with Ach/Acl

        // act = silu(g) * u  ->  bf16 hi/lo into LDS (chunk-local columns)
#pragma unroll
        for (int mt = 0; mt < 2; ++mt)
#pragma unroll
            for (int nt = 0; nt < 2; ++nt)
#pragma unroll
                for (int r = 0; r < 8; ++r) {
                    float g = Cg[mt][nt][r];
                    float u = Cu[mt][nt][r];
                    float a = (g / (1.0f + __expf(-g))) * u;
                    int m = mt * 16 + r + half * 8;
                    int c = wv * 32 + nt * 16 + ln;
                    __bf16 h = f2bf(a);
                    Ach[m * AS + c] = h;
                    Acl[m * AS + c] = f2bf(a - bf2f(h));
                }
        __syncthreads();

        // ---- phase B: Y += act * Wd  (wave owns H columns [wv*128, wv*128+128)) ----
#pragma unroll 1
        for (int kt2 = 0; kt2 < IC / 32; ++kt2) {
            const int k0 = kt2 * 32;
            v16bf a_h[2], a_l[2];
#pragma unroll
            for (int mt = 0; mt < 2; ++mt) {
                a_h[mt] = load_a(Ach, AS, mt * 16, k0, lane);
                a_l[mt] = load_a(Acl, AS, mt * 16, k0, lane);
            }
            // tile row index inside the [E*ID][HD] swizzled matrix
            const size_t ktg = (size_t)(e * (ID / 32) + ci * (IC / 32) + kt2);
#pragma unroll
            for (int nt = 0; nt < 8; ++nt) {
                const size_t tile = ktg * (HD / 16) + (wv * 8 + nt);
                v16bf bh = load_swz(wd_h, tile, lane);
                v16bf bl = load_swz(wd_l, tile, lane);
                Y[0][nt] = wmma3(a_h[0], a_l[0], bh, bl, Y[0][nt]);
                Y[1][nt] = wmma3(a_h[1], a_l[1], bh, bl, Y[1][nt]);
            }
        }
    }

    // epilogue: scale by routing weight, combine across the token's 2 experts
#pragma unroll
    for (int mt = 0; mt < 2; ++mt)
#pragma unroll
        for (int nt = 0; nt < 8; ++nt)
#pragma unroll
            for (int r = 0; r < 8; ++r) {
                int m = mt * 16 + r + half * 8;
                int n = wv * 128 + nt * 16 + ln;
                float val = Y[mt][nt][r] * s_wgt[m];
                atomicAdd(out + (size_t)s_tok[m] * HD + n, val);
            }
}

// ---------- host launch --------------------------------------------------------
extern "C" void kernel_launch(void* const* d_in, const int* in_sizes, int n_in,
                              void* d_out, int out_size, void* d_ws, size_t ws_size,
                              hipStream_t stream) {
    const float* hs = (const float*)d_in[0];   // [T, H]
    const float* rw = (const float*)d_in[1];   // [H, E]
    const float* wg = (const float*)d_in[2];   // [E, H, I]
    const float* wu = (const float*)d_in[3];   // [E, H, I]
    const float* wd = (const float*)d_in[4];   // [E, I, H]

    float* out    = (float*)d_out;                       // [T, H]
    float* logits = out + (size_t)T_TOK * HD;            // [T, E]
    float* sel    = logits + (size_t)T_TOK * NE;         // [T, 2] (as float)

    char* ws = (char*)d_ws;
    int*   expert_cnt = (int*)ws;                                        // [E]
    int*   tok_sel    = (int*)(ws + 256);                                // [T*2]
    float* tok_wt     = (float*)(ws + 256 + (size_t)T_TOK * 2 * 4);      // [T*2]
    int*   expert_tok = (int*)(ws + 256 + (size_t)T_TOK * 2 * 8);        // [E*T]
    float* expert_wt  = (float*)(ws + 256 + (size_t)T_TOK * 2 * 8
                                        + (size_t)NE * T_TOK * 4);       // [E*T]

    // swizzled bf16 hi/lo weights: 6 arrays of E*H*I elems (64 MB each)
    const size_t WELEMS = (size_t)NE * HD * ID;          // 33,554,432
    __bf16* swz = (__bf16*)(ws + 384 * 1024);
    __bf16* wg_h = swz;
    __bf16* wg_l = wg_h + WELEMS;
    __bf16* wu_h = wg_l + WELEMS;
    __bf16* wu_l = wu_h + WELEMS;
    __bf16* wd_h = wu_l + WELEMS;
    __bf16* wd_l = wd_h + WELEMS;

    // ---- weight convert+swizzle pre-pass (memory bound, runs once per call) ----
    {
        // w_gate / w_up as [E*HD][ID]: tiles = (E*HD/32) * (ID/16)
        const int tilesA = (NE * HD / 32) * (ID / 16);   // 65536
        moe_swz_kernel<<<tilesA / 8, 256, 0, stream>>>(wg, wg_h, wg_l, ID, ID / 16, tilesA);
        moe_swz_kernel<<<tilesA / 8, 256, 0, stream>>>(wu, wu_h, wu_l, ID, ID / 16, tilesA);
        // w_down as [E*ID][HD]: tiles = (E*ID/32) * (HD/16)
        const int tilesD = (NE * ID / 32) * (HD / 16);   // 65536
        moe_swz_kernel<<<tilesD / 8, 256, 0, stream>>>(wd, wd_h, wd_l, HD, HD / 16, tilesD);
    }

    const int n4 = T_TOK * HD / 4;
    moe_zero_kernel<<<(n4 + 255) / 256, 256, 0, stream>>>((float4*)out, n4);

    moe_router_kernel<<<T_TOK / 8, 256, 0, stream>>>(hs, rw, logits, sel, tok_wt, tok_sel);

    moe_compact_kernel<<<1, NE, 0, stream>>>(tok_sel, tok_wt, expert_tok, expert_wt, expert_cnt);

    moe_ffn_kernel<<<NE * (T_TOK / MBLK), 256, 0, stream>>>(
        hs, wg_h, wg_l, wu_h, wu_l, wd_h, wd_l,
        expert_cnt, expert_tok, expert_wt, out);
}
